// GRUModel_16784732192778
// MI455X (gfx1250) — matrix operand order repaired and checked by reading, not compiled, not run on hardware
//
#include <hip/hip_runtime.h>

#define HDIM 768
#define LSEQ 4096
#define PSEQ 1024
#define NBLK 16
#define NTHR 256
#define JPW  48        // hidden units per workgroup (16 WGs * 48 = 768)
#define NTILE 9        // 3 gates * 3 sixteen-row tiles (144 rows)
#define NCHUNK 24      // 768 / 32 K-chunks
#define NSTEP (LSEQ - 1 + PSEQ)   // 5119 recurrent steps

typedef _Float16 v16h __attribute__((ext_vector_type(16)));
typedef float    v8f  __attribute__((ext_vector_type(8)));

// ---- dynamic LDS layout (bytes) ----
#define OFF_WHH   0
#define SZ_WHH    (NTILE * NCHUNK * 512 * 2)     // 221184: W_hh slice, f16, WMMA-A swizzled
#define OFF_FO1   (OFF_WHH + SZ_WHH)
#define SZ_FO1    (2 * NCHUNK * 512 * 2)         // 49152: fo1_W, f16, WMMA-A swizzled
#define OFF_HF16  (OFF_FO1 + SZ_FO1)             // 768 * 2
#define OFF_HF32  (OFF_HF16 + HDIM * 2)          // 768 * 4
#define OFF_GH    (OFF_HF32 + HDIM * 4)          // 144 * 4
#define OFF_U     (OFF_GH + 144 * 4)             // 32 * 4
#define OFF_A32   (OFF_U + 32 * 4)               // 32 * 4
#define OFF_WIH   (OFF_A32 + 32 * 4)             // 144*3*4
#define OFF_BIH   (OFF_WIH + 144 * 3 * 4)        // 144*4
#define OFF_BHH   (OFF_BIH + 144 * 4)            // 144*4
#define OFF_FO1B  (OFF_BHH + 144 * 4)            // 32*4
#define OFF_FO2W  (OFF_FO1B + 32 * 4)            // 96*4
#define OFF_FO2B  (OFF_FO2W + 96 * 4)            // 32 (3 floats + pad)
#define OFF_YP    (OFF_FO2B + 32)                // 32 (3 floats + pad)
#define SMEM_TOTAL (OFF_YP + 32)                 // ~279 KB < 320 KB/WGP

__device__ __forceinline__ float sigm(float v) { return 1.0f / (1.0f + __expf(-v)); }

__global__ __launch_bounds__(NTHR, 1)
void gru_rollout_kernel(
    const float* __restrict__ x, const float* __restrict__ w0,
    const float* __restrict__ pred_t,
    const float* __restrict__ W_ih, const float* __restrict__ W_hh,
    const float* __restrict__ b_ih, const float* __restrict__ b_hh,
    const float* __restrict__ fc0_W, const float* __restrict__ fc0_b,
    const float* __restrict__ fc01_W, const float* __restrict__ fc01_b,
    const float* __restrict__ fo1_W, const float* __restrict__ fo1_b,
    const float* __restrict__ fo2_W, const float* __restrict__ fo2_b,
    float* __restrict__ out,
    float* __restrict__ hbuf,          // 2 * 768 floats (ping-pong)
    unsigned* __restrict__ barcnt)     // grid barrier counter (reset per launch)
{
  extern __shared__ char smem[];
  _Float16* whh_s  = (_Float16*)(smem + OFF_WHH);
  _Float16* fo1_s  = (_Float16*)(smem + OFF_FO1);
  _Float16* hf16   = (_Float16*)(smem + OFF_HF16);
  float* hf32   = (float*)(smem + OFF_HF32);
  float* ghb    = (float*)(smem + OFF_GH);
  float* ubuf   = (float*)(smem + OFF_U);
  float* a32    = (float*)(smem + OFF_A32);
  float* wih_s  = (float*)(smem + OFF_WIH);
  float* bih_s  = (float*)(smem + OFF_BIH);
  float* bhh_s  = (float*)(smem + OFF_BHH);
  float* fo1b_s = (float*)(smem + OFF_FO1B);
  float* fo2w_s = (float*)(smem + OFF_FO2W);
  float* fo2b_s = (float*)(smem + OFF_FO2B);
  float* ypArr  = (float*)(smem + OFF_YP);

  const int tid  = threadIdx.x;
  const int wg   = blockIdx.x;
  const int lane = tid & 31;          // wave32
  const int wave = tid >> 5;          // 8 waves per block
  const int half = lane >> 4;

  // ================= init: swizzle resident weights into WMMA A layout ==========
  // A fragment (16x32 f16): lane<16 holds row M=lane; element i (0..15) maps to
  // K = i + (half<<4)  within the 32-wide K chunk.
  for (int e = tid; e < NTILE * NCHUNK * 512; e += NTHR) {
    int blk = e >> 9, r = e & 511;
    int ln = r >> 4, i = r & 15;
    int tile = blk / NCHUNK, c = blk % NCHUNK;
    int gate = tile / 3;
    int row = gate * HDIM + wg * JPW + (tile % 3) * 16 + (ln & 15);
    int k = c * 32 + i + ((ln >> 4) << 4);
    whh_s[e] = (_Float16)W_hh[row * HDIM + k];
  }
  for (int e = tid; e < 2 * NCHUNK * 512; e += NTHR) {
    int blk = e >> 9, r = e & 511;
    int ln = r >> 4, i = r & 15;
    int tile = blk / NCHUNK, c = blk % NCHUNK;
    int row = tile * 16 + (ln & 15);
    int k = c * 32 + i + ((ln >> 4) << 4);
    fo1_s[e] = (_Float16)fo1_W[row * HDIM + k];
  }
  for (int i = tid; i < 144; i += NTHR) {
    int gate = i / JPW, j = i % JPW;
    int row = gate * HDIM + wg * JPW + j;
    bih_s[i] = b_ih[row];
    bhh_s[i] = b_hh[row];
    wih_s[i * 3 + 0] = W_ih[row * 3 + 0];
    wih_s[i * 3 + 1] = W_ih[row * 3 + 1];
    wih_s[i * 3 + 2] = W_ih[row * 3 + 2];
  }
  if (tid < 32) fo1b_s[tid] = fo1_b[tid];
  if (tid < 96) fo2w_s[tid] = fo2_W[tid];
  if (tid < 3)  fo2b_s[tid] = fo2_b[tid];

  // ================= h0 = fc01_W @ relu(fc0_W @ w0 + fc0_b) + fc01_b ============
  if (tid < 32) {
    float acc = fc0_b[tid];
    #pragma unroll
    for (int k = 0; k < 3; ++k) acc = fmaf(fc0_W[tid * 3 + k], w0[k], acc);
    a32[tid] = fmaxf(acc, 0.0f);
  }
  __syncthreads();
  if (tid < JPW) {
    int row = wg * JPW + tid;
    float acc = fc01_b[row];
    #pragma unroll
    for (int k = 0; k < 32; ++k) acc = fmaf(fc01_W[row * 32 + k], a32[k], acc);
    hbuf[row] = acc;                       // buffer 0 = h0
  }
  if (wg == 0 && tid < 3) out[tid] = x[1 + tid];   // y0 = x[0,0,1:4]

  // ================= grid-wide barrier (16 blocks, atomic counter) ==============
  unsigned bar = 0;
  auto gridbar = [&]() {
    __threadfence();
    __syncthreads();
    if (tid == 0) {
      atomicAdd(barcnt, 1u);
      unsigned tgt = (bar + 1u) * (unsigned)NBLK;
      while (atomicAdd(barcnt, 0u) < tgt) __builtin_amdgcn_s_sleep(1);
    }
    bar++;
    __syncthreads();
    __threadfence();
  };
  gridbar();   // h0 visible device-wide

  const v16h* Bp = (const v16h*)hf16;   // B fragment source (h broadcast over N)

  // ================= 5119 sequential GRU steps =================================
  for (int s = 0; s < NSTEP; ++s) {
    const bool obs = (s < LSEQ - 1);
    float p0, p1, p2, dt;
    if (obs) {
      p0 = x[s * 4 + 1]; p1 = x[s * 4 + 2]; p2 = x[s * 4 + 3];
      dt = x[(s + 1) * 4] - x[s * 4];
    } else {
      int pi = s - (LSEQ - 1);
      p0 = ypArr[0]; p1 = ypArr[1]; p2 = ypArr[2];
      float tp = (pi == 0) ? x[(LSEQ - 1) * 4] : pred_t[pi - 1];
      dt = pred_t[pi] - tp;
    }

    // ---- stage h(s) into LDS (f32 for gate math, f16 for WMMA B) ----
    const float* hr = hbuf + (s & 1) * HDIM;
    __syncthreads();
    for (int i = tid; i < HDIM; i += NTHR) {
      float v = hr[i];
      hf32[i] = v;
      hf16[i] = (_Float16)v;
    }
    __syncthreads();

    // ---- gh = W_hh_slice @ h via v_wmma_f32_16x16x32_f16, f32 accumulation ----
    for (int tile = wave; tile < NTILE; tile += 8) {
      v8f acc = {};
      const _Float16* Ab = whh_s + tile * (NCHUNK * 512) + lane * 16;
      #pragma unroll
      for (int c = 0; c < NCHUNK; ++c) {
        v16h a = *(const v16h*)(Ab + c * 512);
        v16h b = Bp[c * 2 + half];          // h chunk, replicated across N columns
        acc = __builtin_amdgcn_wmma_f32_16x16x32_f16(
                  false, a, false, b, (short)0, acc, false, false);
      }
      if ((lane & 15) == 0) {               // N=0 column: lane 0 -> M 0..7, lane 16 -> M 8..15
        int mb = tile * 16 + half * 8;
        #pragma unroll
        for (int r2 = 0; r2 < 8; ++r2) ghb[mb + r2] = acc[r2];
      }
    }
    __syncthreads();

    // ---- GRU gate math for this WG's 48 hidden units -> h(s+1) slice ----
    float* hw = hbuf + ((s + 1) & 1) * HDIM;
    if (tid < JPW) {
      int j = tid, jz = 48 + tid, jn = 96 + tid;
      float gr  = fmaf(wih_s[j * 3 + 0], p0, fmaf(wih_s[j * 3 + 1], p1,
                  fmaf(wih_s[j * 3 + 2], p2, bih_s[j]))) + ghb[j] + bhh_s[j];
      float gz  = fmaf(wih_s[jz * 3 + 0], p0, fmaf(wih_s[jz * 3 + 1], p1,
                  fmaf(wih_s[jz * 3 + 2], p2, bih_s[jz]))) + ghb[jz] + bhh_s[jz];
      float inn = fmaf(wih_s[jn * 3 + 0], p0, fmaf(wih_s[jn * 3 + 1], p1,
                  fmaf(wih_s[jn * 3 + 2], p2, bih_s[jn])));
      float hn_ = ghb[jn] + bhh_s[jn];
      float rg = sigm(gr), zg = sigm(gz);
      float ng = tanhf(inn + rg * hn_);
      float hold = hf32[wg * JPW + j];
      hw[wg * JPW + j] = (1.0f - zg) * ng + zg * hold;
    }

    gridbar();   // h(s+1) complete device-wide (ping-pong keeps h(s) intact)

    // ---- stage h(s+1) for velocity MLP ----
    for (int i = tid; i < HDIM; i += NTHR) {
      float v = hw[i];
      hf32[i] = v;
      hf16[i] = (_Float16)v;
    }
    __syncthreads();

    // ---- u = relu(fo1_W @ h + fo1_b): 2 WMMA tiles on waves 0..1 (redundant
    //      per-WG so prediction feedback needs no second grid barrier) ----
    if (wave < 2) {
      v8f acc = {};
      const _Float16* Ab = fo1_s + wave * (NCHUNK * 512) + lane * 16;
      #pragma unroll
      for (int c = 0; c < NCHUNK; ++c) {
        v16h a = *(const v16h*)(Ab + c * 512);
        v16h b = Bp[c * 2 + half];
        acc = __builtin_amdgcn_wmma_f32_16x16x32_f16(
                  false, a, false, b, (short)0, acc, false, false);
      }
      if ((lane & 15) == 0) {
        int mb = wave * 16 + half * 8;
        #pragma unroll
        for (int r2 = 0; r2 < 8; ++r2) {
          float u = acc[r2] + fo1b_s[mb + r2];
          ubuf[mb + r2] = fmaxf(u, 0.0f);
        }
      }
    }
    __syncthreads();

    // ---- yp = pos + (fo2_W @ u + fo2_b) * dt ----
    if (tid < 3) {
      float v = fo2b_s[tid];
      #pragma unroll
      for (int k = 0; k < 32; ++k) v = fmaf(fo2w_s[tid * 32 + k], ubuf[k], v);
      float pos = (tid == 0) ? p0 : (tid == 1) ? p1 : p2;
      float yp = pos + v * dt;
      ypArr[tid] = yp;
      if (wg == 0) out[(1 + s) * 3 + tid] = yp;
    }
    __syncthreads();
  }
}

extern "C" void kernel_launch(void* const* d_in, const int* in_sizes, int n_in,
                              void* d_out, int out_size, void* d_ws, size_t ws_size,
                              hipStream_t stream) {
  const float* x      = (const float*)d_in[0];
  const float* w0     = (const float*)d_in[1];
  const float* pred_t = (const float*)d_in[2];
  const float* W_ih   = (const float*)d_in[3];
  const float* W_hh   = (const float*)d_in[4];
  const float* b_ih   = (const float*)d_in[5];
  const float* b_hh   = (const float*)d_in[6];
  const float* fc0_W  = (const float*)d_in[7];
  const float* fc0_b  = (const float*)d_in[8];
  const float* fc01_W = (const float*)d_in[9];
  const float* fc01_b = (const float*)d_in[10];
  const float* fo1_W  = (const float*)d_in[11];
  const float* fo1_b  = (const float*)d_in[12];
  const float* fo2_W  = (const float*)d_in[13];
  const float* fo2_b  = (const float*)d_in[14];
  float* out = (float*)d_out;

  unsigned* barcnt = (unsigned*)d_ws;
  float* hbuf = (float*)((char*)d_ws + 256);

  // reset grid-barrier counter every launch (capture-legal)
  hipMemsetAsync(d_ws, 0, 256, stream);

  // allow ~279 KB dynamic LDS (WGP has 320 KB)
  hipFuncSetAttribute(reinterpret_cast<const void*>(gru_rollout_kernel),
                      hipFuncAttributeMaxDynamicSharedMemorySize, SMEM_TOTAL);

  gru_rollout_kernel<<<NBLK, NTHR, SMEM_TOTAL, stream>>>(
      x, w0, pred_t, W_ih, W_hh, b_ih, b_hh,
      fc0_W, fc0_b, fc01_W, fc01_b, fo1_W, fo1_b, fo2_W, fo2_b,
      out, hbuf, barcnt);
}
